// SupConLoss_25924422599109
// MI455X (gfx1250) — matrix operand order repaired and checked by reading, not compiled, not run on hardware
//
#include <hip/hip_runtime.h>
#include <math.h>

typedef __attribute__((ext_vector_type(16))) _Float16 v16h;
typedef __attribute__((ext_vector_type(8)))  _Float16 v8h;
typedef __attribute__((ext_vector_type(8)))  float    v8f;
typedef __attribute__((ext_vector_type(4)))  int      v4i;

#define BTR      1024
#define TDIM     8192
#define NFREQ    819      // T//10
#define NPAD     832      // padded to multiple of 16
#define KPSD     1638     // 819*2
#define KPSD_PAD 1664
#define DFEAT    256
#define NP       3
#define EPSV     1e-8f
#define PI2      6.28318530717958647692f

#define AS1 __attribute__((address_space(1)))
#define AS3 __attribute__((address_space(3)))

// Async global->LDS copy of one 16B chunk (CDNA5 GLOBAL_LOAD_ASYNC_TO_LDS_B128,
// tracked by ASYNCcnt). Builtin signature (probe-confirmed via diagnostics):
//   void (v4i __device__* src, v4i __shared__* dst, imm int offset, imm int cpol)
#if __has_builtin(__builtin_amdgcn_global_load_async_to_lds_b128)
#define ASYNC_CP16(l, g) __builtin_amdgcn_global_load_async_to_lds_b128( \
        (AS1 v4i*)(v4i*)(_Float16*)(g), (AS3 v4i*)(v4i*)(l), 0, 0)
#else
#define ASYNC_CP16(l, g) (*(v8h*)(l) = *(const v8h*)(g))
#endif

#if __has_builtin(__builtin_amdgcn_s_wait_asynccnt)
#define ASYNC_WAIT() __builtin_amdgcn_s_wait_asynccnt(0)
#else
#define ASYNC_WAIT() asm volatile("s_wait_asynccnt 0" ::: "memory")
#endif

#define LDP 40   // LDS row pitch in halfs (80B): 16B aligned, spreads banks

// ---------------------------------------------------------------------------
// LDS-tiled C(MxN) = A(MxK) * B(NxK)^T, f16 in / f32 accum.
// Block tile 128(M) x 64(N), 8 waves; wave w owns rows [16w,16w+16) x all 64
// columns (4 WMMA subtiles -> 4x A-reuse, 8x B-reuse vs per-wave tiles).
// Per K-step: cooperatively async-stage A(128x32) + B(64x32) into LDS,
// s_wait_asynccnt + barrier, then 4 chained v_wmma_f32_16x16x32_f16 per wave.
// All tile indices are uniform -> EXEC all-ones across every WMMA (ISA 7.12).
// ---------------------------------------------------------------------------
__global__ __launch_bounds__(256)
void gemm_nt_f16(const _Float16* __restrict__ A, const _Float16* __restrict__ B,
                 float* __restrict__ C, int M, int N, int K,
                 int lda, int ldb, int ldc)
{
    __shared__ __align__(16) _Float16 sA[128 * LDP];
    __shared__ __align__(16) _Float16 sB[64 * LDP];

    const int tid    = threadIdx.x;
    const int lane   = tid & 31;
    const int wave   = tid >> 5;
    const int lrow   = lane & 15;
    const int hi     = lane >> 4;              // 0 = low half, 1 = high half
    const int tilesN = N >> 6;                 // 64-wide block tiles in N
    const int bm     = blockIdx.x / tilesN;
    const int bn     = blockIdx.x % tilesN;
    const int m0     = bm << 7;                // 128-row block tile
    const int n0     = bn << 6;

    // Cooperative staging map: A = 512 16B chunks (2/thread), B = 256 (1/thread)
    const int ar0 = tid >> 1;                  // A rows come in thread pairs
    const int ac0 = (tid & 1) << 4;            // first of two 8-half chunks
    const int br  = tid >> 2;
    const int bc  = (tid & 3) << 3;

    const _Float16* agp = A + (size_t)(m0 + ar0) * lda + ac0;
    const _Float16* bgp = B + (size_t)(n0 + br)  * ldb + bc;
    _Float16* alp0 = sA + ar0 * LDP + ac0;
    _Float16* blp  = sB + br  * LDP + bc;

    // Fragment source pointers in LDS (ISA 7.12.2 wave32 layouts):
    //  A 16x32: low lanes K 0-7 & 16-23, high lanes K 8-15 & 24-31
    //  B 32x16 (from NT rows): low lanes K 0-7 & 16-23, high lanes K 8-15 & 24-31
    const _Float16* la = sA + (wave * 16 + lrow) * LDP + hi * 8;

    v8f acc[4] = {};
    for (int k0 = 0; k0 < K; k0 += 32) {
        ASYNC_CP16(alp0,     agp + k0);
        ASYNC_CP16(alp0 + 8, agp + k0 + 8);
        ASYNC_CP16(blp,      bgp + k0);
        ASYNC_WAIT();
        __syncthreads();

        union { v16h v; v8h h[2]; } af;
        af.h[0] = *(const v8h*)(la);
        af.h[1] = *(const v8h*)(la + 16);
#pragma unroll
        for (int s = 0; s < 4; ++s) {
            union { v16h v; v8h h[2]; } bf;
            const _Float16* lb = sB + (s * 16 + lrow) * LDP + hi * 8;
            bf.h[0] = *(const v8h*)(lb);
            bf.h[1] = *(const v8h*)(lb + 16);
            acc[s] = __builtin_amdgcn_wmma_f32_16x16x32_f16(
                         false, af.v, false, bf.v, (short)0, acc[s], false, false);
        }
        __syncthreads();
    }
    // D layout: lane, vgpr r -> row = m0 + wave*16 + r + 8*hi, col = n0 + s*16 + lrow
#pragma unroll
    for (int s = 0; s < 4; ++s) {
        float* cp = C + (size_t)(m0 + wave * 16 + hi * 8) * ldc + n0 + s * 16 + lrow;
#pragma unroll
        for (int r = 0; r < 8; ++r) cp[(size_t)r * ldc] = acc[s][r];
    }
}

// ---------------------------------------------------------------------------
// Stream trajectories once (the 128MB HBM-bound pass): emit f16 V matrix
// (rows 2b,2b+1 = velocity channels 1,3), per-channel mean, and winding
// numbers (sum of diffs telescopes to endpoint difference).
// ---------------------------------------------------------------------------
__global__ __launch_bounds__(256)
void prep_traj(const float* __restrict__ traj, _Float16* __restrict__ V,
               float* __restrict__ v_mean, float* __restrict__ winding)
{
    __shared__ float s1[256], s3[256];
    const int b  = blockIdx.x;
    const int t0 = threadIdx.x;
    const float4* row = (const float4*)(traj + (size_t)b * TDIM * 4);
    _Float16* v0 = V + (size_t)(2 * b)     * TDIM;
    _Float16* v1 = V + (size_t)(2 * b + 1) * TDIM;
    float sum1 = 0.f, sum3 = 0.f;
    for (int t = t0; t < TDIM; t += 256) {
        float4 q = row[t];
        v0[t] = (_Float16)q.y;
        v1[t] = (_Float16)q.w;
        sum1 += q.y;
        sum3 += q.w;
    }
    s1[t0] = sum1; s3[t0] = sum3;
    __syncthreads();
    for (int s = 128; s > 0; s >>= 1) {
        if (t0 < s) { s1[t0] += s1[t0 + s]; s3[t0] += s3[t0 + s]; }
        __syncthreads();
    }
    if (t0 == 0) {
        v_mean[2 * b]     = s1[0] * (1.0f / TDIM);
        v_mean[2 * b + 1] = s3[0] * (1.0f / TDIM);
        float4 qa = row[0], qb = row[TDIM - 1];
        winding[2 * b]     = fabsf(rintf((qb.x - qa.x) * (1.0f / PI2)));
        winding[2 * b + 1] = fabsf(rintf((qb.z - qa.z) * (1.0f / PI2)));
    }
}

// Hann-windowed DFT twiddle matrices (rows >= NFREQ zeroed for padding).
__global__ __launch_bounds__(256)
void twiddles(_Float16* __restrict__ Wr, _Float16* __restrict__ Wi)
{
    int idx = blockIdx.x * 256 + threadIdx.x;      // over NPAD*TDIM
    int k = idx >> 13;                             // / TDIM
    int t = idx & (TDIM - 1);
    float wr = 0.f, wi = 0.f;
    if (k < NFREQ) {
        float w  = 0.5f * (1.0f - __cosf(PI2 * (float)t * (1.0f / TDIM)));
        float ph = PI2 * (float)((k * t) & (TDIM - 1)) * (1.0f / TDIM);
        wr =  __cosf(ph) * w;
        wi = -__sinf(ph) * w;
    }
    Wr[idx] = (_Float16)wr;
    Wi[idx] = (_Float16)wi;
}

// psd[b][2k+c] = Re^2 + Im^2 for k < NFREQ, pad region zero.
__global__ __launch_bounds__(256)
void psd_build(const float* __restrict__ Re, const float* __restrict__ Im,
               float* __restrict__ psd, _Float16* __restrict__ psdh)
{
    int idx = blockIdx.x * 256 + threadIdx.x;      // over BTR*KPSD_PAD
    int b = idx / KPSD_PAD;
    int j = idx - b * KPSD_PAD;
    float p = 0.f;
    if (j < KPSD) {
        int k = j >> 1, c = j & 1;
        size_t m = (size_t)(2 * b + c) * NPAD + k;
        float re = Re[m], im = Im[m];
        p = re * re + im * im;
    }
    psd[idx]  = p;
    psdh[idx] = (_Float16)p;
}

__global__ __launch_bounds__(256)
void n2_kernel(const float* __restrict__ psd, float* __restrict__ n2)
{
    __shared__ float s[256];
    int b = blockIdx.x, t = threadIdx.x;
    float a = 0.f;
    for (int j = t; j < KPSD_PAD; j += 256) {
        float p = psd[(size_t)b * KPSD_PAD + j];
        a += p * p;
    }
    s[t] = a;
    __syncthreads();
    for (int sh = 128; sh > 0; sh >>= 1) {
        if (t < sh) s[t] += s[t + sh];
        __syncthreads();
    }
    if (t == 0) n2[b] = s[0];
}

__global__ __launch_bounds__(256)
void feat_norm(const float* __restrict__ F, _Float16* __restrict__ fh)
{
    __shared__ float s[256];
    int i = blockIdx.x, c = threadIdx.x;
    float v = F[(size_t)i * DFEAT + c];
    s[c] = v * v;
    __syncthreads();
    for (int sh = 128; sh > 0; sh >>= 1) {
        if (c < sh) s[c] += s[c + sh];
        __syncthreads();
    }
    float nrm = fmaxf(sqrtf(s[0]), 1e-12f);
    fh[(size_t)i * DFEAT + c] = (_Float16)(v / nrm);
}

__global__ void init_out(float* out) { out[0] = 0.0f; }

// Final fused SupCon reduction. Reference takes triu(total,1)+triu^T, so the
// asymmetric psd_sim row-norm uses row = min(i,j); diagonal contributes 0.
__global__ __launch_bounds__(256)
void supcon_reduce(const float* __restrict__ sim, const float* __restrict__ gram,
                   const float* __restrict__ n2, const float* __restrict__ v_mean,
                   const float* __restrict__ winding, const float* __restrict__ labels,
                   const float* __restrict__ scales, float* __restrict__ out)
{
    __shared__ float sp[256], sd[256];
    const int i = blockIdx.x, t = threadIdx.x;
    const float inv_s = 1.0f / scales[0];
    const float li0 = labels[i * NP], li1 = labels[i * NP + 1], li2 = labels[i * NP + 2];
    const float vmi0 = v_mean[2 * i], vmi1 = v_mean[2 * i + 1];
    const float wi0 = winding[2 * i], wi1 = winding[2 * i + 1];
    const float n2i = n2[i];
    float pos = 0.f, den = 0.f;
    for (int j = t; j < BTR; j += 256) {
        if (j == i) continue;
        float e = __expf(sim[(size_t)i * BTR + j] * 10.0f);   // / TEMPERATURE
        float d0 = (li0 - labels[j * NP])     * inv_s;
        float d1 = (li1 - labels[j * NP + 1]) * inv_s;
        float d2 = (li2 - labels[j * NP + 2]) * inv_s;
        float pd = sqrtf(d0 * d0 + d1 * d1 + d2 * d2);
        float dv0 = vmi0 - v_mean[2 * j], dv1 = vmi1 - v_mean[2 * j + 1];
        float vsim = __expf(-sqrtf(dv0 * dv0 + dv1 * dv1) * 2.0f);  // / 0.5
        float n2j = n2[j];
        float g   = gram[(size_t)i * BTR + j];
        float dd  = fmaxf(n2i + n2j - 2.0f * g, 0.0f);
        float rn2 = (i < j) ? n2i : n2j;                      // triu row norm
        float psim = __expf(-sqrtf(dd) / (sqrtf(rn2) + EPSV));
        float wsim = (wi0 == winding[2 * j] && wi1 == winding[2 * j + 1]) ? 1.0f : 0.0f;
        float kin  = 0.4f * vsim + 0.4f * psim + 0.2f * wsim;
        float mask = __expf(-pd * 10.0f) * kin;               // exp(-dist/0.1)
        pos += e * mask;
        den += e;
    }
    sp[t] = pos; sd[t] = den;
    __syncthreads();
    for (int sh = 128; sh > 0; sh >>= 1) {
        if (t < sh) { sp[t] += sp[t + sh]; sd[t] += sd[t + sh]; }
        __syncthreads();
    }
    if (t == 0) {
        float loss = -__logf((sp[0] + EPSV) / (sd[0] + EPSV));
        atomicAdd(out, loss * (1.0f / BTR));
    }
}

extern "C" void kernel_launch(void* const* d_in, const int* in_sizes, int n_in,
                              void* d_out, int out_size, void* d_ws, size_t ws_size,
                              hipStream_t stream)
{
    const float* features = (const float*)d_in[0];   // (1024, 256)
    const float* labels   = (const float*)d_in[1];   // (1024, 3)
    const float* traj     = (const float*)d_in[2];   // (1024, 8192, 4)
    const float* scales   = (const float*)d_in[3];   // (1,)

    char* ws = (char*)d_ws;
    size_t off = 0;
    auto alloc = [&](size_t bytes) -> char* {
        char* p = ws + off;
        off = (off + bytes + 255) & ~(size_t)255;
        return p;
    };
    _Float16* V    = (_Float16*)alloc((size_t)2 * BTR * TDIM * sizeof(_Float16)); // 32 MB
    _Float16* Wr   = (_Float16*)alloc((size_t)NPAD * TDIM * sizeof(_Float16));    // 13.6 MB
    _Float16* Wi   = (_Float16*)alloc((size_t)NPAD * TDIM * sizeof(_Float16));    // 13.6 MB
    float*    Re   = (float*)   alloc((size_t)2 * BTR * NPAD * sizeof(float));    // 6.8 MB
    float*    Im   = (float*)   alloc((size_t)2 * BTR * NPAD * sizeof(float));    // 6.8 MB
    float*    psd  = (float*)   alloc((size_t)BTR * KPSD_PAD * sizeof(float));    // 6.8 MB
    _Float16* psdh = (_Float16*)alloc((size_t)BTR * KPSD_PAD * sizeof(_Float16)); // 3.4 MB
    float*    gram = (float*)   alloc((size_t)BTR * BTR * sizeof(float));         // 4 MB
    _Float16* fh   = (_Float16*)alloc((size_t)BTR * DFEAT * sizeof(_Float16));
    float*    simm = (float*)   alloc((size_t)BTR * BTR * sizeof(float));         // 4 MB
    float*    vme  = (float*)   alloc((size_t)BTR * 2 * sizeof(float));
    float*    wnd  = (float*)   alloc((size_t)BTR * 2 * sizeof(float));
    float*    n2   = (float*)   alloc((size_t)BTR * sizeof(float));

    // 1) stream trajectories (HBM-bound pass), build V + stats
    prep_traj<<<BTR, 256, 0, stream>>>(traj, V, vme, wnd);
    // 2) windowed-DFT twiddle matrices
    twiddles<<<(NPAD * TDIM) / 256, 256, 0, stream>>>(Wr, Wi);
    // 3) truncated DFT as LDS-tiled WMMA GEMMs: (2048 x 8192) x (8192 x 832)
    {
        int grid = (2 * BTR / 128) * (NPAD / 64);   // 16 x 13 = 208 blocks
        gemm_nt_f16<<<grid, 256, 0, stream>>>(V, Wr, Re, 2 * BTR, NPAD, TDIM, TDIM, TDIM, NPAD);
        gemm_nt_f16<<<grid, 256, 0, stream>>>(V, Wi, Im, 2 * BTR, NPAD, TDIM, TDIM, TDIM, NPAD);
    }
    // 4) power spectrum + norms
    psd_build<<<(BTR * KPSD_PAD) / 256, 256, 0, stream>>>(Re, Im, psd, psdh);
    n2_kernel<<<BTR, 256, 0, stream>>>(psd, n2);
    // 5) psd gram: (1024 x 1664) x (1664 x 1024) via WMMA
    {
        int grid = (BTR / 128) * (BTR / 64);        // 8 x 16 = 128 blocks
        gemm_nt_f16<<<grid, 256, 0, stream>>>(psdh, psdh, gram, BTR, BTR, KPSD_PAD, KPSD_PAD, KPSD_PAD, BTR);
    }
    // 6) normalized features + similarity gram via WMMA
    feat_norm<<<BTR, 256, 0, stream>>>(features, fh);
    {
        int grid = (BTR / 128) * (BTR / 64);
        gemm_nt_f16<<<grid, 256, 0, stream>>>(fh, fh, simm, BTR, BTR, DFEAT, DFEAT, DFEAT, BTR);
    }
    // 7) fused SupCon reduction -> scalar loss
    init_out<<<1, 1, 0, stream>>>((float*)d_out);
    supcon_reduce<<<BTR, 256, 0, stream>>>(simm, gram, n2, vme, wnd, labels, scales, (float*)d_out);
}
